// AttentionWithKVCache_36687610643005
// MI455X (gfx1250) — matrix operand order repaired and checked by reading, not compiled, not run on hardware
//
#include <hip/hip_runtime.h>

typedef __bf16 bf16_t;
typedef bf16_t v16bf __attribute__((ext_vector_type(16)));
typedef float  v8f   __attribute__((ext_vector_type(8)));
typedef unsigned int v4u __attribute__((ext_vector_type(4)));
typedef int v8i __attribute__((ext_vector_type(8)));
typedef int v4i __attribute__((ext_vector_type(4)));
typedef __attribute__((address_space(3))) float lds_float;   // true DS-path pointer

#define SQ   4
#define NH   32
#define NHKV 8
#define GQ   4          // NH / NHKV
#define DK   128
#define DV   128
#define ROWSTRIDE 1024  // NHKV * DK (elements between consecutive seq positions)
#define LDSROW    129   // padded LDS row stride (words) -> bank-conflict-free gather
#define LDSBUF    (32 * LDSROW)   // words per V tile buffer

// ---- TDM: DMA one 32x128 fp32 V tile (16KB) into LDS, padded rows (stride 129 DW) ----
__device__ __forceinline__ void tdm_load_v_tile(const float* gptr, unsigned lds_byte_addr) {
  unsigned long long ga = (unsigned long long)(uintptr_t)gptr;
  // D# group0: count=1 | lds_addr | global_addr[56:0] | type=2
  v4u g0 = { 1u,
             lds_byte_addr,
             (unsigned)(ga & 0xffffffffu),
             (unsigned)((ga >> 32) & 0x01ffffffu) | 0x80000000u };
  // D# group1: data_size=4B(2<<16), pad_enable(1<<20), pad_interval=128DW(6<<22),
  //            pad_amount=1DW(0<<25); tensor_dim0=128; tensor_dim1=32;
  //            tile_dim0=128; tile_dim1=32; tile_dim2=0; dim0_stride=1024
  v8i g1 = { (int)((2u << 16) | (1u << 20) | (6u << 22)),
             (int)(128u << 16),   // tensor_dim0 low16 -> bits 63:48
             (int)(32u  << 16),   // tensor_dim0 hi16=0; tensor_dim1 low16 -> bits 95:80
             (int)(128u << 16),   // tensor_dim1 hi16=0; tile_dim0 -> bits 127:112
             32,                  // tile_dim1=32; tile_dim2=0
             1024,                // tensor_dim0_stride (low32)
             0, 0 };
  v4i gz = {0, 0, 0, 0};
#if defined(__clang_major__) && (__clang_major__ >= 23)
  v8i gz8 = {0, 0, 0, 0, 0, 0, 0, 0};
  __builtin_amdgcn_tensor_load_to_lds(g0, g1, gz, gz, gz8, 0);
#else
  __builtin_amdgcn_tensor_load_to_lds(g0, g1, gz, gz, 0);
#endif
}

// One wave32 per workgroup; one (b, hkv, split) triple per workgroup.
__global__ __launch_bounds__(32) void attn_partial_kernel(
    const float* __restrict__ Q, const float* __restrict__ K,
    const float* __restrict__ V, const int* __restrict__ lens,
    float* __restrict__ pacc, float* __restrict__ pml,
    int Smax, int nsplit, int chunk)
{
  __shared__ float smem[2 * LDSBUF];

  const int wg    = blockIdx.x;
  const int split = wg % nsplit;
  const int bh    = wg / nsplit;      // b*NHKV + hkv
  const int hkv   = bh % NHKV;
  const int b     = bh / NHKV;

  const int lane = threadIdx.x;
  const int n    = lane & 15;         // query column (S^T) / d-column (PV out)
  const int hi   = lane >> 4;

  const int hg    = n >> 2;
  const int m_seq = n & 3;
  const int hq    = hkv * GQ + hg;

  const float scale = 0.08838834764831845f; // 1/sqrt(128)

  // ---- Q row, pre-scaled, in B-operand layout for S^T WMMAs ----
  v16bf qb[4];
  {
    const int qbase = ((b * SQ + m_seq) * NH + hq) * DK;
    #pragma unroll
    for (int c = 0; c < 4; ++c)
      #pragma unroll
      for (int e = 0; e < 16; ++e)
        qb[c][e] = (bf16_t)(Q[qbase + c * 32 + hi * 8 + (e & 7) + (e >> 3) * 16] * scale);
  }

  const int len   = lens[b];
  const int start = split * chunk;                  // 32-aligned
  const int end   = min(start + chunk, len);
  const int limit = len - SQ + m_seq;

  float m_run = -3.0e38f;
  float l_run = 0.0f;
  v8f acc[8];
  #pragma unroll
  for (int c = 0; c < 8; ++c)
    #pragma unroll
    for (int r = 0; r < 8; ++r) acc[c][r] = 0.0f;

  const int kbase = (b * Smax * NHKV + hkv) * DK + n * ROWSTRIDE;   // K row n base
  const float* vtile0 = V + (size_t)(b * Smax * NHKV + hkv) * DV;   // + kb*ROWSTRIDE
  const unsigned lds_base = (unsigned)(uintptr_t)&smem[0];          // LDS byte offset

  // prologue: prefetch first V tile into buffer 0
  if (start < end) tdm_load_v_tile(vtile0 + (size_t)start * ROWSTRIDE, lds_base);

  int buf = 0;
  for (int kb = start; kb < end; kb += 32) {
    const int nkb = kb + 32;
    // make sure LDS reads of the buffer we are about to overwrite have retired
    asm volatile("s_wait_dscnt 0x0" ::: "memory");
    if (nkb < end)
      tdm_load_v_tile(vtile0 + (size_t)nkb * ROWSTRIDE,
                      lds_base + (unsigned)((buf ^ 1) * LDSBUF * 4));

    // ---- S^T tiles: A = K rows (16 keys x 32 d), B = Q^T (register path) ----
    v8f s0, s1;
    #pragma unroll
    for (int r = 0; r < 8; ++r) { s0[r] = 0.0f; s1[r] = 0.0f; }
    const int ka = kbase + kb * ROWSTRIDE;
    #pragma unroll
    for (int cq = 0; cq < 4; ++cq) {
      v16bf a0, a1;
      #pragma unroll
      for (int e = 0; e < 8; ++e) {
        a0[e]     = (bf16_t)K[ka + cq * 32 + hi * 8 + e];
        a0[e + 8] = (bf16_t)K[ka + cq * 32 + 16 + hi * 8 + e];
        a1[e]     = (bf16_t)K[ka + 16 * ROWSTRIDE + cq * 32 + hi * 8 + e];
        a1[e + 8] = (bf16_t)K[ka + 16 * ROWSTRIDE + cq * 32 + 16 + hi * 8 + e];
      }
      s0 = __builtin_amdgcn_wmma_f32_16x16x32_bf16(false, a0, false, qb[cq],
                                                   (short)0, s0, false, false);
      s1 = __builtin_amdgcn_wmma_f32_16x16x32_bf16(false, a1, false, qb[cq],
                                                   (short)0, s1, false, false);
    }

    // ---- mask + online softmax (column n = query; C reg r holds key r+8*hi) ----
    float mx = -3.0e38f;
    #pragma unroll
    for (int r = 0; r < 8; ++r) {
      const int k0 = kb + r + 8 * hi;
      const int k1 = k0 + 16;
      s0[r] = (k0 > limit || k0 >= end) ? -3.0e38f : s0[r];
      s1[r] = (k1 > limit || k1 >= end) ? -3.0e38f : s1[r];
      mx = fmaxf(mx, fmaxf(s0[r], s1[r]));
    }
    mx = fmaxf(mx, __shfl_xor(mx, 16, 32));
    const float m_new = fmaxf(m_run, mx);
    const float scl   = __expf(m_run - m_new);

    // P = exp(s - m_new); C layout of (s0,s1) IS the A layout of P for PV WMMA.
    float ps = 0.0f;
    v16bf pa;
    #pragma unroll
    for (int r = 0; r < 8; ++r) {
      const float p0 = __expf(s0[r] - m_new);
      const float p1 = __expf(s1[r] - m_new);
      ps += p0 + p1;
      pa[r]     = (bf16_t)p0;
      pa[r + 8] = (bf16_t)p1;
    }
    ps += __shfl_xor(ps, 16, 32);
    l_run = l_run * scl + ps;
    m_run = m_new;

    // rescale accumulator rows (out row q = r + 8*hi)
    #pragma unroll
    for (int r = 0; r < 8; ++r) {
      const float srow = __shfl(scl, r + 8 * hi, 32);
      #pragma unroll
      for (int c = 0; c < 8; ++c) acc[c][r] *= srow;
    }

    // ---- wait for current V tile from TDM (uniform branch, imm operand) ----
    if (nkb < end) {
      __builtin_amdgcn_s_wait_tensorcnt(1);   // keep the just-issued prefetch in flight
    } else {
      __builtin_amdgcn_s_wait_tensorcnt(0);   // last iteration: drain
    }
    asm volatile("" ::: "memory");            // no DS read may hoist above the wait

    // addrspace(3) pointer -> true ds_load_b32 gathers (bank-conflict-free, stride 129)
    const lds_float* vs =
        (const lds_float*)(lds_base + (unsigned)(buf * LDSBUF * 4));
    #pragma unroll
    for (int c = 0; c < 8; ++c) {
      v16bf vbop;
      #pragma unroll
      for (int e = 0; e < 16; ++e)
        vbop[e] = (bf16_t)vs[(hi * 8 + (e & 7) + (e >> 3) * 16) * LDSROW + c * 16 + n];
      acc[c] = __builtin_amdgcn_wmma_f32_16x16x32_bf16(false, pa, false, vbop,
                                                       (short)0, acc[c], false, false);
    }
    buf ^= 1;
  }

  // ---- write partials ----
  float* ml = pml + wg * 32;
  if (hi == 0) {
    ml[n * 2 + 0] = m_run;
    ml[n * 2 + 1] = l_run;
  }
  float* pw = pacc + (size_t)wg * 16 * DV;
  #pragma unroll
  for (int c = 0; c < 8; ++c)
    #pragma unroll
    for (int r = 0; r < 8; ++r)
      pw[(r + 8 * hi) * DV + c * 16 + n] = acc[c][r];
}

// Combine split-K partials: one block per (b, hkv).
__global__ __launch_bounds__(256) void attn_combine_kernel(
    const float* __restrict__ pacc, const float* __restrict__ pml,
    float* __restrict__ out, int nsplit)
{
  const int bh  = blockIdx.x;
  const int hkv = bh % NHKV;
  const int b   = bh / NHKV;
  for (int idx = threadIdx.x; idx < 16 * DV; idx += 256) {
    const int q = idx >> 7;
    const int d = idx & (DV - 1);
    const float* mlq = pml + (size_t)(bh * nsplit) * 32 + q * 2;
    float M = -3.0e38f;
    for (int s = 0; s < nsplit; ++s) M = fmaxf(M, mlq[s * 32]);
    float L = 0.0f, A = 0.0f;
    const float* pa = pacc + (size_t)(bh * nsplit) * 16 * DV + q * DV + d;
    for (int s = 0; s < nsplit; ++s) {
      const float w = __expf(mlq[s * 32] - M);
      L += mlq[s * 32 + 1] * w;
      A += pa[(size_t)s * 16 * DV] * w;
    }
    const int hg = q >> 2, m_seq = q & 3;
    const int h  = hkv * GQ + hg;
    out[((b * SQ + m_seq) * NH + h) * DV + d] = A / (L > 0.0f ? L : 1.0f);
  }
}

extern "C" void kernel_launch(void* const* d_in, const int* in_sizes, int n_in,
                              void* d_out, int out_size, void* d_ws, size_t ws_size,
                              hipStream_t stream) {
  const float* Q    = (const float*)d_in[0];
  const float* K    = (const float*)d_in[1];
  const float* V    = (const float*)d_in[2];
  const int*   lens = (const int*)d_in[3];
  float* out = (float*)d_out;

  const int B    = in_sizes[3];
  const int Smax = (int)((long long)in_sizes[1] / ((long long)B * NHKV * DK));

  const size_t per_split = (size_t)B * NHKV * 16 * (DV + 2) * sizeof(float);
  int nsplit = (int)(ws_size / per_split);
  if (nsplit > 32) nsplit = 32;
  if (nsplit < 1)  nsplit = 1;
  int chunk = ((Smax + nsplit - 1) / nsplit + 31) & ~31;   // 32-aligned slice

  float* pacc = (float*)d_ws;
  float* pml  = pacc + (size_t)B * NHKV * nsplit * 16 * DV;

  attn_partial_kernel<<<dim3(B * NHKV * nsplit), dim3(32), 0, stream>>>(
      Q, K, V, lens, pacc, pml, Smax, nsplit, chunk);
  attn_combine_kernel<<<dim3(B * NHKV), dim3(256), 0, stream>>>(
      pacc, pml, out, nsplit);
}
